// FlashAttention2_13932873908514
// MI455X (gfx1250) — matrix operand order repaired and checked by reading, not compiled, not run on hardware
//
#include <hip/hip_runtime.h>

// ---------------------------------------------------------------------------
// Problem constants (match reference)
// ---------------------------------------------------------------------------
#define HIDDEN   2048
#define NH       16
#define NKV      4
#define HD       128             // head dim
#define BATCH    2
#define SEQ      2048
#define ROWS     (BATCH * SEQ)   // 4096
#define KVW      (NKV * HD)      // 512

typedef _Float16 half_t;
typedef __attribute__((ext_vector_type(16))) _Float16     v16h;
typedef __attribute__((ext_vector_type(8)))  float        v8f;
typedef __attribute__((ext_vector_type(4)))  unsigned int v4u;
typedef __attribute__((ext_vector_type(8)))  int          v8i;
typedef __attribute__((ext_vector_type(4)))  int          v4i;

// ---------------------------------------------------------------------------
// WMMA helper: D = A(16x32 f16) * B(32x16 f16) + C(16x16 f32)
// ---------------------------------------------------------------------------
__device__ __forceinline__ v8f wmma_f16(v16h a, v16h b, v8f c) {
  return __builtin_amdgcn_wmma_f32_16x16x32_f16(
      false, a, false, b, (short)0, c, false, false);
}

// A-fragment (16x32 f16). Element (m, k) at base[m*ld + k].
// this code (contiguous k): lanes 0-15 -> M=lane, K=0..15; lanes 16-31 -> K=16..31
__device__ __forceinline__ v16h load_frag_a(const half_t* base, int ld, int k0, int lane) {
  const int m  = lane & 15;
  const int kh = (lane >> 4) << 4;
  const half_t* row = base + (size_t)m * ld + k0;
  v16h f;
#pragma unroll
  for (int v = 0; v < 8; ++v) {
    const int kk = kh + (v << 1);
    f[2 * v]     = row[kk];
    f[2 * v + 1] = row[kk + 1];
  }
  return f;
}

// B-fragment (32x16 f16). Element (k, n) at base[n*ld + k] (k contiguous per column).
// this code (contiguous k): lanes 0-15 -> N=lane, K=0..15; lanes 16-31 -> K=16..31
__device__ __forceinline__ v16h load_frag_b(const half_t* base, int ld, int k0, int lane) {
  const int n  = lane & 15;
  const int kh = (lane >> 4) << 4;
  const half_t* col = base + (size_t)n * ld + k0 + kh;
  v16h f;
#pragma unroll
  for (int v = 0; v < 8; ++v) {
    f[2 * v]     = col[2 * v];
    f[2 * v + 1] = col[2 * v + 1];
  }
  return f;
}

// ---------------------------------------------------------------------------
// TDM: async 2D tile load Global -> LDS via Tensor Data Mover.
// Descriptor packed per CDNA5 ISA section 8 (D# groups 0/1); 2D tile == tensor
// view, data_size = 2 bytes, LDS row padding via pad_interval/pad_amount.
//   dim0   : tile width in halves (x, contiguous)
//   dim1   : tile height (y)
//   stride0: row stride in halves
//   pad_i  : pad_interval code (0=2 dw,1=4,...,5=64,...)
//   pad_a  : pad_amount code (0=1 dw,...,3=4 dw)
// ---------------------------------------------------------------------------
__device__ __forceinline__ void tdm_load_2d(unsigned lds_addr, unsigned long long gaddr,
                                            unsigned dim0, unsigned dim1,
                                            unsigned long long stride0,
                                            unsigned pad_i, unsigned pad_a) {
  v4u g0;
  g0[0] = 1u;                                             // count=1, user descriptor
  g0[1] = lds_addr;                                       // LDS byte offset
  g0[2] = (unsigned)(gaddr & 0xFFFFFFFFull);              // global_addr[31:0]
  g0[3] = (unsigned)((gaddr >> 32) & 0x01FFFFFFull)       // global_addr[56:32]
        | (2u << 30);                                     // type = 2 ("image")
  v8i g1;
  g1[0] = (int)((1u << 16)                                // data_size = 2 bytes
              | (1u << 20)                                // pad_enable
              | (pad_i << 22) | (pad_a << 25));
  g1[1] = (int)((dim0 & 0xFFFFu) << 16);                  // tensor_dim0 lo16 (abar=0)
  g1[2] = (int)(((dim0 >> 16) & 0xFFFFu) | ((dim1 & 0xFFFFu) << 16)); // dim0 hi | dim1 lo
  g1[3] = (int)(((dim1 >> 16) & 0xFFFFu) | ((dim0 & 0xFFFFu) << 16)); // dim1 hi | tile_dim0
  g1[4] = (int)(dim1 & 0xFFFFu);                          // tile_dim1 | tile_dim2=0
  g1[5] = (int)(stride0 & 0xFFFFFFFFull);                 // tensor_dim0_stride lo32
  g1[6] = (int)((stride0 >> 32) & 0xFFFFull);             // stride hi16 | dim1_stride lo16=0
  g1[7] = 0;
  const v4i z4 = {0, 0, 0, 0};
#if defined(__clang_major__) && (__clang_major__ >= 23)
  const v8i z8 = {0, 0, 0, 0, 0, 0, 0, 0};
  __builtin_amdgcn_tensor_load_to_lds(g0, g1, z4, z4, z8, 0);
#else
  __builtin_amdgcn_tensor_load_to_lds(g0, g1, z4, z4, 0);
#endif
}

__device__ __forceinline__ unsigned lds_off(const void* p) {
  // generic address of LDS object: low 32 bits == byte offset in workgroup LDS
  return (unsigned)(unsigned long long)(uintptr_t)p;
}

// ---------------------------------------------------------------------------
// fp32 -> f16 converter (n multiple of 4)
// ---------------------------------------------------------------------------
__global__ void cvt_f32_f16_kernel(const float* __restrict__ in,
                                   half_t* __restrict__ out, int n) {
  int i = (blockIdx.x * blockDim.x + threadIdx.x) << 2;
  if (i + 3 < n) {
    float4 v = *reinterpret_cast<const float4*>(in + i);
    out[i]     = (half_t)v.x;
    out[i + 1] = (half_t)v.y;
    out[i + 2] = (half_t)v.z;
    out[i + 3] = (half_t)v.w;
  }
}

// ---------------------------------------------------------------------------
// fp32 [R][C] -> f16 transposed [C][R]  (LDS-tiled; R,C multiples of 32)
// ---------------------------------------------------------------------------
__global__ __launch_bounds__(256)
void transpose_f32_f16_kernel(const float* __restrict__ in, half_t* __restrict__ out,
                              int R, int C) {
  __shared__ float t[32][33];
  const int c0 = blockIdx.x * 32, r0 = blockIdx.y * 32;
  const int tx = threadIdx.x, ty0 = threadIdx.y;
#pragma unroll
  for (int j = 0; j < 4; ++j) {
    const int ty = ty0 + 8 * j;
    t[ty][tx] = in[(size_t)(r0 + ty) * C + c0 + tx];
  }
  __syncthreads();
#pragma unroll
  for (int j = 0; j < 4; ++j) {
    const int ty = ty0 + 8 * j;
    out[(size_t)(c0 + ty) * R + r0 + tx] = (half_t)t[tx][ty];
  }
}

// ---------------------------------------------------------------------------
// f16 batched transpose: Vr [hb][SEQ][HD] -> Vt [hb][HD][SEQ]
// ---------------------------------------------------------------------------
__global__ __launch_bounds__(256)
void transpose_f16_kernel(const half_t* __restrict__ in, half_t* __restrict__ out) {
  __shared__ half_t t[32][33];
  const int s0 = blockIdx.x * 32, d0 = blockIdx.y * 32;
  const size_t base = (size_t)blockIdx.z * SEQ * HD;
  const int tx = threadIdx.x, ty0 = threadIdx.y;
#pragma unroll
  for (int j = 0; j < 4; ++j) {
    const int ty = ty0 + 8 * j;
    t[ty][tx] = in[base + (size_t)(s0 + ty) * HD + d0 + tx];
  }
  __syncthreads();
#pragma unroll
  for (int j = 0; j < 4; ++j) {
    const int ty = ty0 + 8 * j;
    out[base + (size_t)(d0 + ty) * SEQ + s0 + tx] = t[tx][ty];
  }
}

// ---------------------------------------------------------------------------
// WMMA GEMM: C[M,N] = A[M,K](f16 row-major) @ Bt[N,K](f16, pre-transposed) + bias
//   block tile 128x128, 8 waves, wave tile 32x64 (2x4 WMMAs), K-step 32.
//   M%128==0, N%128==0, K%32==0.
// ---------------------------------------------------------------------------
#define GLDA 40   // padded LDS row (halves)

__global__ __launch_bounds__(256)
void wmma_gemm_kernel(const half_t* __restrict__ A, const half_t* __restrict__ Bt,
                      const float* __restrict__ bias, float* __restrict__ C,
                      int N, int K) {
  __shared__ half_t sA[128][GLDA];
  __shared__ half_t sB[128][GLDA];   // sB[n][k]

  const int tid  = threadIdx.x;
  const int lane = tid & 31;
  const int w    = tid >> 5;
  const int wm   = w & 3;            // 4 row groups of 32
  const int wn   = w >> 2;           // 2 col groups of 64
  const int rowBase = blockIdx.y * 128;
  const int colBase = blockIdx.x * 128;

  v8f cc[2][4];
  const v8f vzero = {};
#pragma unroll
  for (int i = 0; i < 2; ++i)
#pragma unroll
    for (int j = 0; j < 4; ++j) cc[i][j] = vzero;

  const int nkt = K >> 5;
  for (int kt = 0; kt < nkt; ++kt) {
    const int k0 = kt << 5;
    // stage A and B tiles (each 128x32 halves = 1024 uint2 chunks)
#pragma unroll
    for (int i = 0; i < 4; ++i) {
      const int c  = tid + (i << 8);
      const int r  = c >> 3;
      const int kk = (c & 7) << 2;
      *reinterpret_cast<uint2*>(&sA[r][kk]) =
          *reinterpret_cast<const uint2*>(A  + (size_t)(rowBase + r) * K + k0 + kk);
      *reinterpret_cast<uint2*>(&sB[r][kk]) =
          *reinterpret_cast<const uint2*>(Bt + (size_t)(colBase + r) * K + k0 + kk);
    }
    __syncthreads();

    if (kt + 1 < nkt) {   // hint next tiles into cache
      __builtin_prefetch(A  + (size_t)(rowBase + (tid >> 1)) * K + k0 + 32, 0, 1);
      __builtin_prefetch(Bt + (size_t)(colBase + (tid >> 1)) * K + k0 + 32, 0, 1);
    }

    const v16h a0 = load_frag_a(&sA[wm * 32][0],      GLDA, 0, lane);
    const v16h a1 = load_frag_a(&sA[wm * 32 + 16][0], GLDA, 0, lane);
    v16h bf[4];
#pragma unroll
    for (int j = 0; j < 4; ++j)
      bf[j] = load_frag_b(&sB[wn * 64 + j * 16][0], GLDA, 0, lane);
#pragma unroll
    for (int j = 0; j < 4; ++j) {
      cc[0][j] = wmma_f16(a0, bf[j], cc[0][j]);
      cc[1][j] = wmma_f16(a1, bf[j], cc[1][j]);
    }
    __syncthreads();
  }

  const int hlf = lane >> 4;
  const int nn  = lane & 15;
#pragma unroll
  for (int i = 0; i < 2; ++i)
#pragma unroll
    for (int j = 0; j < 4; ++j) {
      const int n0 = colBase + wn * 64 + j * 16 + nn;
      const float bn = bias[n0];
#pragma unroll
      for (int r = 0; r < 8; ++r) {
        const int m0 = rowBase + wm * 32 + i * 16 + r + 8 * hlf;
        C[(size_t)m0 * N + n0] = cc[i][j][r] + bn;
      }
    }
}

// ---------------------------------------------------------------------------
// RoPE + head split + f16 cast.
// ---------------------------------------------------------------------------
__global__ void rope_kernel(const float* __restrict__ Qlin, const float* __restrict__ Klin,
                            const float* __restrict__ Vlin, const int* __restrict__ pos_ids,
                            half_t* __restrict__ Qr, half_t* __restrict__ Kr,
                            half_t* __restrict__ Vr) {
  const int rs = blockIdx.x;             // b*S + s
  const int b  = rs >> 11;
  const int s  = rs & (SEQ - 1);
  const int d  = threadIdx.x;            // 0..127
  const int i  = d & 63;

  const float pos  = (float)pos_ids[rs];
  const float invf = __expf(-(float)i * (13.815510557964274f / 64.0f)); // theta^(-i/64)
  const float fr   = pos * invf;
  const float cs   = cosf(fr);
  const float sn   = sinf(fr);

  const float* qrow = Qlin + (size_t)rs * HIDDEN;
#pragma unroll
  for (int h = 0; h < NH; ++h) {
    const float x   = qrow[h * HD + d];
    const float xp  = qrow[h * HD + (d ^ 64)];
    const float rot = (d < 64) ? -xp : xp;
    Qr[((size_t)(b * NH + h) * SEQ + s) * HD + d] = (half_t)(x * cs + rot * sn);
  }
  const float* krow = Klin + (size_t)rs * KVW;
#pragma unroll
  for (int h = 0; h < NKV; ++h) {
    const float x   = krow[h * HD + d];
    const float xp  = krow[h * HD + (d ^ 64)];
    const float rot = (d < 64) ? -xp : xp;
    Kr[((size_t)(b * NKV + h) * SEQ + s) * HD + d] = (half_t)(x * cs + rot * sn);
  }
  const float* vrow = Vlin + (size_t)rs * KVW;
#pragma unroll
  for (int h = 0; h < NKV; ++h)
    Vr[((size_t)(b * NKV + h) * SEQ + s) * HD + d] = (half_t)vrow[h * HD + d];
}

// ---------------------------------------------------------------------------
// Causal GQA flash attention; K/V tiles staged by the Tensor Data Mover.
//   grid = (S/128, B*NH); block 256 (8 waves); wave owns 16 query rows.
//   V is pre-transposed: Vt [hb][HD][SEQ].
// ---------------------------------------------------------------------------
#define KLD 136   // 128 + 8 pad halves (pad = 4 dwords every 64 dwords)
#define VLD 72    // 64 + 8 pad halves  (pad = 4 dwords every 32 dwords)
#define PLD 72

__global__ __launch_bounds__(256)
void flash_attn_kernel(const half_t* __restrict__ Q, const half_t* __restrict__ Kc,
                       const half_t* __restrict__ Vt, half_t* __restrict__ Ctx) {
  __shared__ __align__(16) half_t sK[64][KLD];     // [key][d]
  __shared__ __align__(16) half_t sVt[HD][VLD];    // [d][key]
  __shared__ __align__(16) half_t sP[8][16][PLD];  // wave-private probs

  const int tid  = threadIdx.x;
  const int lane = tid & 31;
  const int w    = tid >> 5;
  const int hlf  = lane >> 4;
  const int nn   = lane & 15;

  const int hb  = blockIdx.y;            // 0..B*NH-1
  const int b   = hb >> 4;
  const int h   = hb & 15;
  const int kvh = h >> 2;                // GQA

  const int qb = blockIdx.x * 128;
  const int qw = qb + w * 16;

  const half_t* qbase  = Q  + ((size_t)hb * SEQ + qw) * HD;
  const half_t* kbase  = Kc + (size_t)(b * NKV + kvh) * SEQ * HD;
  const half_t* vtbase = Vt + (size_t)(b * NKV + kvh) * SEQ * HD;
  const unsigned sK_off  = lds_off(&sK[0][0]);
  const unsigned sVt_off = lds_off(&sVt[0][0]);

  v16h qf[4];
#pragma unroll
  for (int dc = 0; dc < 4; ++dc) qf[dc] = load_frag_a(qbase, HD, dc * 32, lane);

  float mrow[8], lrow[8];
  v8f   o[8];
  const v8f vzero = {};
#pragma unroll
  for (int r = 0; r < 8; ++r) { mrow[r] = -1e30f; lrow[r] = 0.0f; }
#pragma unroll
  for (int t = 0; t < 8; ++t) o[t] = vzero;

  const int ntiles = (qb >> 6) + 2;      // 64-key tiles through the diagonal
  for (int kt = 0; kt < ntiles; ++kt) {
    const int kb = kt << 6;
    // ---- TDM staging: wave 0 issues both tile DMAs, waits TENSORcnt==0
    if (w == 0) {
      // K tile: 64 rows x 128 halves, row stride HD
      tdm_load_2d(sK_off, (unsigned long long)(uintptr_t)(kbase + (size_t)kb * HD),
                  128u, 64u, (unsigned long long)HD, 5u, 3u);
      // V tile (transposed source): 128 rows(d) x 64 halves(s), row stride SEQ
      tdm_load_2d(sVt_off, (unsigned long long)(uintptr_t)(vtbase + kb),
                  64u, 128u, (unsigned long long)SEQ, 4u, 3u);
      __builtin_amdgcn_s_wait_tensorcnt(0);
    }
    __syncthreads();

    // ---- scores: wave's 16 rows x 64 keys
    v8f sc[4];
#pragma unroll
    for (int nk = 0; nk < 4; ++nk) {
      v8f acc = {};
#pragma unroll
      for (int dc = 0; dc < 4; ++dc)
        acc = wmma_f16(qf[dc], load_frag_b(&sK[nk * 16][0], KLD, dc * 32, lane), acc);
      sc[nk] = acc;
    }

    // ---- scale + causal mask
    const float scale = 0.08838834764831845f;   // 1/sqrt(128)
#pragma unroll
    for (int nk = 0; nk < 4; ++nk) {
      const int kidx = kb + nk * 16 + nn;
#pragma unroll
      for (int r = 0; r < 8; ++r) {
        const int qidx = qw + r + 8 * hlf;
        const float v  = sc[nk][r] * scale;
        sc[nk][r] = (kidx <= qidx) ? v : -1e30f;
      }
    }

    // ---- online softmax (row reductions across 16-lane halves)
#pragma unroll
    for (int r = 0; r < 8; ++r) {
      float mloc = fmaxf(fmaxf(sc[0][r], sc[1][r]), fmaxf(sc[2][r], sc[3][r]));
#pragma unroll
      for (int off = 8; off >= 1; off >>= 1)
        mloc = fmaxf(mloc, __shfl_xor(mloc, off, 32));
      const float mnew = fmaxf(mrow[r], mloc);
      const float corr = __expf(mrow[r] - mnew);
      float rs = 0.0f;
#pragma unroll
      for (int nk = 0; nk < 4; ++nk) {
        const float p = __expf(sc[nk][r] - mnew);
        sc[nk][r] = p;
        rs += p;
      }
#pragma unroll
      for (int off = 8; off >= 1; off >>= 1) rs += __shfl_xor(rs, off, 32);
      lrow[r] = lrow[r] * corr + rs;
      mrow[r] = mnew;
#pragma unroll
      for (int t = 0; t < 8; ++t) o[t][r] *= corr;
    }

    // ---- C-layout -> A-layout via wave-private LDS
#pragma unroll
    for (int nk = 0; nk < 4; ++nk)
#pragma unroll
      for (int r = 0; r < 8; ++r)
        sP[w][r + 8 * hlf][nk * 16 + nn] = (half_t)sc[nk][r];
    asm volatile("s_wait_dscnt 0" ::: "memory");   // wave-local LDS RAW fence

    // ---- P @ V
#pragma unroll
    for (int kc = 0; kc < 2; ++kc) {
      const v16h pa = load_frag_a(&sP[w][0][0], PLD, kc * 32, lane);
#pragma unroll
      for (int t = 0; t < 8; ++t)
        o[t] = wmma_f16(pa, load_frag_b(&sVt[t * 16][0], VLD, kc * 32, lane), o[t]);
    }
    __syncthreads();   // protect sK/sVt before next TDM overwrite
  }

  // ---- epilogue: Ctx[b, s, h*128 + d] = o / l  (f16)
#pragma unroll
  for (int t = 0; t < 8; ++t) {
    const int d = t * 16 + nn;
#pragma unroll
    for (int r = 0; r < 8; ++r) {
      const int qrow = qw + r + 8 * hlf;
      Ctx[((size_t)(b * SEQ + qrow)) * HIDDEN + h * HD + d] =
          (half_t)(o[t][r] / lrow[r]);
    }
  }
}

// ---------------------------------------------------------------------------
// Host-side launch sequence
// ---------------------------------------------------------------------------
extern "C" void kernel_launch(void* const* d_in, const int* in_sizes, int n_in,
                              void* d_out, int out_size, void* d_ws, size_t ws_size,
                              hipStream_t stream) {
  const float* X   = (const float*)d_in[0];
  const int*   pos = (const int*)  d_in[1];
  const float* Wq  = (const float*)d_in[2];
  const float* bq  = (const float*)d_in[3];
  const float* Wk  = (const float*)d_in[4];
  const float* bk  = (const float*)d_in[5];
  const float* Wv  = (const float*)d_in[6];
  const float* bv  = (const float*)d_in[7];
  const float* Wo  = (const float*)d_in[8];
  const float* bo  = (const float*)d_in[9];
  float* out = (float*)d_out;

  char* ws = (char*)d_ws;
  size_t off = 0;
  auto take = [&](size_t bytes) -> void* {
    void* p = ws + off;
    off += (bytes + 255) & ~(size_t)255;
    return p;
  };
  half_t* Xh   = (half_t*)take((size_t)ROWS * HIDDEN * 2);
  half_t* WqT  = (half_t*)take((size_t)HIDDEN * HIDDEN * 2);   // [N][K]
  half_t* WkT  = (half_t*)take((size_t)KVW * HIDDEN * 2);
  half_t* WvT  = (half_t*)take((size_t)KVW * HIDDEN * 2);
  half_t* WoT  = (half_t*)take((size_t)HIDDEN * HIDDEN * 2);
  float*  Qlin = (float*)take((size_t)ROWS * HIDDEN * 4);
  float*  Klin = (float*)take((size_t)ROWS * KVW * 4);
  float*  Vlin = (float*)take((size_t)ROWS * KVW * 4);
  half_t* Qr   = (half_t*)take((size_t)ROWS * HIDDEN * 2);
  half_t* Kr   = (half_t*)take((size_t)ROWS * KVW * 2);
  half_t* Vr   = (half_t*)take((size_t)ROWS * KVW * 2);
  half_t* Vt   = (half_t*)take((size_t)ROWS * KVW * 2);
  half_t* Ctx  = (half_t*)Qlin;   // reuse Qlin region after RoPE

  // 1) cast X, transpose+cast weights
  cvt_f32_f16_kernel<<<(ROWS * HIDDEN / 4 + 255) / 256, 256, 0, stream>>>(X, Xh, ROWS * HIDDEN);
  transpose_f32_f16_kernel<<<dim3(HIDDEN / 32, HIDDEN / 32), dim3(32, 8), 0, stream>>>(Wq, WqT, HIDDEN, HIDDEN);
  transpose_f32_f16_kernel<<<dim3(KVW    / 32, HIDDEN / 32), dim3(32, 8), 0, stream>>>(Wk, WkT, HIDDEN, KVW);
  transpose_f32_f16_kernel<<<dim3(KVW    / 32, HIDDEN / 32), dim3(32, 8), 0, stream>>>(Wv, WvT, HIDDEN, KVW);
  transpose_f32_f16_kernel<<<dim3(HIDDEN / 32, HIDDEN / 32), dim3(32, 8), 0, stream>>>(Wo, WoT, HIDDEN, HIDDEN);

  // 2) QKV projections
  wmma_gemm_kernel<<<dim3(HIDDEN / 128, ROWS / 128), 256, 0, stream>>>(Xh, WqT, bq, Qlin, HIDDEN, HIDDEN);
  wmma_gemm_kernel<<<dim3(KVW    / 128, ROWS / 128), 256, 0, stream>>>(Xh, WkT, bk, Klin, KVW,    HIDDEN);
  wmma_gemm_kernel<<<dim3(KVW    / 128, ROWS / 128), 256, 0, stream>>>(Xh, WvT, bv, Vlin, KVW,    HIDDEN);

  // 3) RoPE + head split; then transpose V per head for TDM-friendly tiles
  rope_kernel<<<ROWS, HD, 0, stream>>>(Qlin, Klin, Vlin, pos, Qr, Kr, Vr);
  transpose_f16_kernel<<<dim3(SEQ / 32, HD / 32, BATCH * NKV), dim3(32, 8), 0, stream>>>(Vr, Vt);

  // 4) causal GQA flash attention (TDM-staged K/V)
  flash_attn_kernel<<<dim3(SEQ / 128, BATCH * NH), 256, 0, stream>>>(Qr, Kr, Vt, Ctx);

  // 5) output projection -> d_out (fp32)
  wmma_gemm_kernel<<<dim3(HIDDEN / 128, ROWS / 128), 256, 0, stream>>>(Ctx, WoT, bo, out, HIDDEN, HIDDEN);
}
